// DiTBlock_56410100465666
// MI455X (gfx1250) — matrix operand order repaired and checked by reading, not compiled, not run on hardware
//
#include <hip/hip_runtime.h>
#include <hip/hip_bf16.h>

#define HIDDEN 1024
#define NHEADS 16
#define HD 64
#define SEQ 2048
#define BATCH 2
#define ROWS (BATCH * SEQ)      // 4096
#define MLPH 4096

typedef __attribute__((ext_vector_type(16))) _Float16 v16h;
typedef __attribute__((ext_vector_type(8)))  _Float16 v8h;
typedef __attribute__((ext_vector_type(8)))  float    v8f;
typedef __attribute__((ext_vector_type(4)))  unsigned int v4u;
typedef __attribute__((ext_vector_type(4)))  int       v4i;
typedef __attribute__((ext_vector_type(8)))  int       v8i;

// ---------------------------------------------------------------------------
// Tensor Data Mover support (gfx1250). 2-D tile load: global -> LDS with row pad.
// ---------------------------------------------------------------------------
#if defined(__has_builtin)
#if __has_builtin(__builtin_amdgcn_tensor_load_to_lds) && __has_builtin(__builtin_amdgcn_s_wait_tensorcnt)
#define USE_TDM 1
#endif
#endif

#ifdef USE_TDM
// Loads a tile of tile_d0 x tile_d1 elements (2 bytes each) from a row-major
// tensor (row length = dim0 elements, row stride = dim0) into LDS at lds_off,
// adding 4 DWORDs (8 halves) of padding after every 16 DWORDs (32 halves).
__device__ __forceinline__ void tdm_load_tile_f16(unsigned lds_off, const void* gptr,
                                                  unsigned dim0, unsigned dim1,
                                                  unsigned tile_d0, unsigned tile_d1) {
  unsigned long long ga = (unsigned long long)(uintptr_t)gptr;
  v4u g0;
  g0[0] = 1u;                                         // count=1, user descriptor
  g0[1] = lds_off;                                    // lds_addr
  g0[2] = (unsigned)(ga & 0xFFFFFFFFu);               // global_addr[31:0]
  g0[3] = (unsigned)((ga >> 32) & 0x1FFFFFFu) | (2u << 30);  // addr[56:32], type=2
  v8i g1;
  g1[0] = (int)((1u << 16) | (1u << 20) | (3u << 22) | (3u << 25));
          // data_size=2B, pad_enable, pad_interval=16 DW, pad_amount=4 DW
  g1[1] = (int)((dim0 & 0xFFFFu) << 16);                              // tensor_dim0 lo16
  g1[2] = (int)(((dim0 >> 16) & 0xFFFFu) | ((dim1 & 0xFFFFu) << 16)); // dim0 hi | dim1 lo
  g1[3] = (int)(((dim1 >> 16) & 0xFFFFu) | ((tile_d0 & 0xFFFFu) << 16)); // dim1 hi | tile_dim0
  g1[4] = (int)(tile_d1 & 0xFFFFu);                                   // tile_dim1 | tile_dim2=0
  g1[5] = (int)dim0;                                                  // tensor_dim0_stride lo32
  g1[6] = 0;
  g1[7] = 0;
  v4i z4 = {};
#if defined(__clang_major__) && (__clang_major__ >= 23)
  v8i z8 = {};
  __builtin_amdgcn_tensor_load_to_lds(g0, g1, z4, z4, z8, 0);
#else
  __builtin_amdgcn_tensor_load_to_lds(g0, g1, z4, z4, 0);
#endif
}
#endif

// ---------------------------------------------------------------------------
// mod = silu(c) @ w_ada + b_ada   (2 x 6144, tiny GEMV -> plain VALU kernel)
// ---------------------------------------------------------------------------
__global__ void k_silu_ada(const float* __restrict__ c, const float* __restrict__ w,
                           const float* __restrict__ bias, float* __restrict__ mod) {
  int j = blockIdx.x * blockDim.x + threadIdx.x;
  if (j >= BATCH * 6 * HIDDEN) return;
  int b = j / (6 * HIDDEN), col = j % (6 * HIDDEN);
  float acc = bias[col];
  for (int i = 0; i < HIDDEN; ++i) {
    float cv = c[b * HIDDEN + i];
    acc += (cv / (1.f + __expf(-cv))) * w[(size_t)i * (6 * HIDDEN) + col];
  }
  mod[j] = acc;
}

// ---------------------------------------------------------------------------
// Weight cast + transpose: out[N][K] f16 = in[K][N] f32
// ---------------------------------------------------------------------------
__global__ void k_cast_transpose(const float* __restrict__ in, _Float16* __restrict__ out,
                                 int K, int N) {
  int i = blockIdx.x * blockDim.x + threadIdx.x;
  if (i >= K * N) return;
  int n = i / K, k = i - n * K;
  out[(size_t)n * K + k] = (_Float16)in[(size_t)k * N + n];
}

// ---------------------------------------------------------------------------
// h = ln(x) * (1 + sc) + sh  -> f16 (one 256-thread block per token row)
// ---------------------------------------------------------------------------
__global__ void k_ln_mod(const float* __restrict__ x, const float* __restrict__ mod,
                         int sh_off, int sc_off, _Float16* __restrict__ out) {
  __shared__ float red[256];
  int row = blockIdx.x;
  int b = row / SEQ;
  const float* xr = x + (size_t)row * HIDDEN;
  float s = 0.f;
  for (int i = threadIdx.x; i < HIDDEN; i += 256) s += xr[i];
  red[threadIdx.x] = s; __syncthreads();
  for (int o = 128; o > 0; o >>= 1) {
    if (threadIdx.x < o) red[threadIdx.x] += red[threadIdx.x + o];
    __syncthreads();
  }
  float mean = red[0] * (1.f / HIDDEN);
  __syncthreads();
  float v = 0.f;
  for (int i = threadIdx.x; i < HIDDEN; i += 256) { float d = xr[i] - mean; v += d * d; }
  red[threadIdx.x] = v; __syncthreads();
  for (int o = 128; o > 0; o >>= 1) {
    if (threadIdx.x < o) red[threadIdx.x] += red[threadIdx.x + o];
    __syncthreads();
  }
  float rs = rsqrtf(red[0] * (1.f / HIDDEN) + 1e-6f);
  const float* shv = mod + (size_t)b * 6 * HIDDEN + sh_off;
  const float* scv = mod + (size_t)b * 6 * HIDDEN + sc_off;
  for (int i = threadIdx.x; i < HIDDEN; i += 256) {
    float hval = (xr[i] - mean) * rs * (1.f + scv[i]) + shv[i];
    out[(size_t)row * HIDDEN + i] = (_Float16)hval;
  }
}

// ---------------------------------------------------------------------------
// Tiled WMMA GEMM: C[M,N] = A[M,K](f16) @ Bt[N,K](f16, pre-transposed) + bias
// Block tile 64x128, 8 waves of 32x32. A tile double-buffered in LDS via the
// Tensor Data Mover (DMA for chunk k+1 overlaps WMMA on chunk k); B fragments
// loaded contiguously from transposed global weights (resident in 192MB L2).
// ---------------------------------------------------------------------------
#define BM 64
#define BN 128
#define BK 32
#define LDA 40    // halves; 80B row stride, multiple of 16B (TDM pad 16DW+4DW)

__global__ __launch_bounds__(256) void k_gemm(const _Float16* __restrict__ A,
                                              const _Float16* __restrict__ Bt,
                                              const float* __restrict__ bias,
                                              float* __restrict__ outF,
                                              _Float16* __restrict__ outH,
                                              int M, int N, int K, int act) {
  __shared__ _Float16 As[2][BM][LDA];
  int bm = blockIdx.y * BM, bn = blockIdx.x * BN;
  int tid = threadIdx.x;
  int lane = tid & 31;
  int w = tid >> 5;
  int wm = (w & 1) * 32, wn = (w >> 1) * 32;
  int g = lane >> 4, ln = lane & 15;
  v8f acc[2][2] = {};
  const _Float16* brow[2];
  brow[0] = Bt + (size_t)(bn + wn + ln) * K;
  brow[1] = Bt + (size_t)(bn + wn + 16 + ln) * K;

#ifdef USE_TDM
  // prologue: kick off DMA for the first A tile
  if (tid < 32) {
    unsigned lds0 = (unsigned)(unsigned long long)(uintptr_t)&As[0][0][0];
    tdm_load_tile_f16(lds0, A + (size_t)bm * K, (unsigned)K, (unsigned)M, BK, BM);
  }
#endif
  int cur = 0;
  for (int k0 = 0; k0 < K; k0 += BK) {
    __syncthreads();   // prior reads of the buffer about to be overwritten done
#ifdef USE_TDM
    if (tid < 32) {
      if (k0 + BK < K) {
        unsigned ldsn = (unsigned)(unsigned long long)(uintptr_t)&As[cur ^ 1][0][0];
        tdm_load_tile_f16(ldsn, A + (size_t)bm * K + k0 + BK,
                          (unsigned)K, (unsigned)M, BK, BM);
        __builtin_amdgcn_s_wait_tensorcnt(1);  // current tile done, next in flight
      } else {
        __builtin_amdgcn_s_wait_tensorcnt(0);
      }
    }
#else
    {
      int r = tid >> 2, cc = (tid & 3) * 8;
      *(v8h*)&As[cur][r][cc] = *(const v8h*)(A + (size_t)(bm + r) * K + k0 + cc);
    }
#endif
    __syncthreads();

    __builtin_prefetch(brow[0] + k0 + 2 * BK, 0, 1);
    __builtin_prefetch(brow[1] + k0 + 2 * BK, 0, 1);

    v16h af[2];
#pragma unroll
    for (int i = 0; i < 2; ++i) {
      int row = wm + i * 16 + ln;
      v8h lo = *(const v8h*)&As[cur][row][g * 16];
      v8h hi = *(const v8h*)&As[cur][row][g * 16 + 8];
#pragma unroll
      for (int e = 0; e < 8; ++e) { af[i][e] = lo[e]; af[i][8 + e] = hi[e]; }
    }
    v16h bf[2];
#pragma unroll
    for (int j = 0; j < 2; ++j)
      bf[j] = *(const v16h*)(brow[j] + k0 + g * 16);

#pragma unroll
    for (int i = 0; i < 2; ++i)
#pragma unroll
      for (int j = 0; j < 2; ++j)
        acc[i][j] = __builtin_amdgcn_wmma_f32_16x16x32_f16(
            false, af[i], false, bf[j], (short)0, acc[i][j], false, false);
    cur ^= 1;
  }

#pragma unroll
  for (int i = 0; i < 2; ++i)
#pragma unroll
    for (int j = 0; j < 2; ++j) {
      int nG = bn + wn + j * 16 + ln;
      float bv = bias[nG];
#pragma unroll
      for (int r = 0; r < 8; ++r) {
        int mG = bm + wm + i * 16 + r + 8 * g;
        float vv = acc[i][j][r] + bv;
        if (act == 1) {
          // tanh-gelu: 0.5*v*(1+tanh(u)) == v * sigmoid(2u)
          float u = 0.7978845608028654f * (vv + 0.044715f * vv * vv * vv);
          vv = vv / (1.f + __expf(-2.f * u));
        }
        size_t idx = (size_t)mG * N + nG;
        if (outH) outH[idx] = (_Float16)vv; else outF[idx] = vv;
      }
    }
}

// ---------------------------------------------------------------------------
// RoPE (first 64 channels, per reference) + split + scale q.
// qh/kh: [B*H, N, 64] f16 ; vt: [B*H, 64, N] f16 (transposed for B-fragments)
// ---------------------------------------------------------------------------
__global__ void k_rope_split(const float* __restrict__ qkv, _Float16* __restrict__ qh,
                             _Float16* __restrict__ kh, _Float16* __restrict__ vt) {
  int idx = blockIdx.x * blockDim.x + threadIdx.x;
  if (idx >= ROWS * HIDDEN) return;
  int row = idx >> 10, c = idx & 1023;
  int b = row / SEQ, n = row % SEQ;
  const float* base = qkv + (size_t)row * 3 * HIDDEN;
  float q = base[c], k = base[HIDDEN + c], v = base[2 * HIDDEN + c];
  if (c < HD) {
    int pi = c >> 1;
    // 10000^(-2i/64) = exp2(-(2i/64)*log2(10000))
    float inv = exp2f(-(2.f * (float)pi) * (13.287712379549449f / (float)HD));
    float ang = (float)n * inv;
    float cs = __cosf(ang), sn = __sinf(ang);
    float qp = base[c ^ 1], kp = base[HIDDEN + (c ^ 1)];
    if ((c & 1) == 0) { q = q * cs - qp * sn; k = k * cs - kp * sn; }
    else              { q = q * cs + qp * sn; k = k * cs + kp * sn; }
  }
  q *= 0.125f;  // 1/sqrt(64) folded into Q
  int h = c >> 6, d = c & 63;
  int bh = b * NHEADS + h;
  size_t o = ((size_t)bh * SEQ + n) * HD + d;
  qh[o] = (_Float16)q;
  kh[o] = (_Float16)k;
  vt[((size_t)bh * HD + d) * SEQ + n] = (_Float16)v;
}

// ---------------------------------------------------------------------------
// Flash attention: 1 wave per (b*head, 16-query tile). WMMA for QK^T and PV.
// ctx: [B, N, HIDDEN] f16
// ---------------------------------------------------------------------------
__global__ __launch_bounds__(32) void k_attn(const _Float16* __restrict__ qh,
                                             const _Float16* __restrict__ kh,
                                             const _Float16* __restrict__ vt,
                                             const float* __restrict__ rel_table,
                                             _Float16* __restrict__ ctx) {
  __shared__ _Float16 P[16][32];
  int bh = blockIdx.x;            // b*16 + head
  int qt = blockIdx.y;            // 16-query tile
  int head = bh & (NHEADS - 1);
  int bb = bh >> 4;
  int lane = threadIdx.x;
  int g = lane >> 4, ln = lane & 15;
  const _Float16* Q  = qh + ((size_t)bh * SEQ + qt * 16) * HD;
  const _Float16* Km = kh + (size_t)bh * SEQ * HD;
  const _Float16* Vt = vt + (size_t)bh * HD * SEQ;

  // Q A-fragments for d 0..31 and 32..63
  v16h qa0, qa1;
  {
    const _Float16* qrow = Q + (size_t)ln * HD;
    v8h l0 = *(const v8h*)(qrow + g * 16);
    v8h h0 = *(const v8h*)(qrow + g * 16 + 8);
    v8h l1 = *(const v8h*)(qrow + 32 + g * 16);
    v8h h1 = *(const v8h*)(qrow + 40 + g * 16);
#pragma unroll
    for (int e = 0; e < 8; ++e) {
      qa0[e] = l0[e]; qa0[8 + e] = h0[e];
      qa1[e] = l1[e]; qa1[8 + e] = h1[e];
    }
  }

  float mstate[8], lstate[8];
  v8f outacc[4] = {};
#pragma unroll
  for (int r = 0; r < 8; ++r) { mstate[r] = -1e30f; lstate[r] = 0.f; }

  for (int kc = 0; kc < SEQ; kc += 32) {
    v8f s[2];
#pragma unroll
    for (int t = 0; t < 2; ++t) {
      int key = kc + t * 16 + ln;           // this lane's key column
      const _Float16* krow = Km + (size_t)key * HD;
      v16h kb0 = *(const v16h*)(krow + g * 16);        // d = g*16 + e
      v16h kb1 = *(const v16h*)(krow + 32 + g * 16);   // d = 32 + g*16 + e
      v8f cc = {};
      cc = __builtin_amdgcn_wmma_f32_16x16x32_f16(false, qa0, false, kb0, (short)0, cc, false, false);
      cc = __builtin_amdgcn_wmma_f32_16x16x32_f16(false, qa1, false, kb1, (short)0, cc, false, false);
      // T5 relative bias (C layout: lane -> key, vgpr r -> row r + 8*g)
#pragma unroll
      for (int r = 0; r < 8; ++r) {
        int qabs = qt * 16 + r + 8 * g;
        int rp = key - qabs;
        int bucket = rp > 0 ? 16 : 0;
        int arp = rp < 0 ? -rp : rp;
        if (arp < 8) bucket += arp;
        else {
          int rpl = 8 + (int)(__logf((float)arp * 0.125f) * (8.0f / 2.772588722239781f)); // /log(16)*8
          bucket += (rpl < 15 ? rpl : 15);
        }
        cc[r] += rel_table[bucket * NHEADS + head];
      }
      s[t] = cc;
    }

    // online softmax: reductions within each 16-lane group
    float alpha[8];
#pragma unroll
    for (int r = 0; r < 8; ++r) {
      float mv = fmaxf(s[0][r], s[1][r]);
      for (int off = 1; off < 16; off <<= 1) mv = fmaxf(mv, __shfl_xor(mv, off, 32));
      float mnew = fmaxf(mstate[r], mv);
      alpha[r] = __expf(mstate[r] - mnew);
      mstate[r] = mnew;
    }
#pragma unroll
    for (int r = 0; r < 8; ++r) {
      float p0 = __expf(s[0][r] - mstate[r]);
      float p1 = __expf(s[1][r] - mstate[r]);
      int m = r + 8 * g;
      P[m][ln]      = (_Float16)p0;
      P[m][16 + ln] = (_Float16)p1;
      float rs = p0 + p1;
      for (int off = 1; off < 16; off <<= 1) rs += __shfl_xor(rs, off, 32);
      lstate[r] = lstate[r] * alpha[r] + rs;
    }
    __syncthreads();

    // P A-fragment (16x32)
    v16h pf;
    {
      const _Float16* prow = &P[ln][0];
      v8h lo = *(const v8h*)(prow + g * 16);
      v8h hi = *(const v8h*)(prow + g * 16 + 8);
#pragma unroll
      for (int e = 0; e < 8; ++e) { pf[e] = lo[e]; pf[8 + e] = hi[e]; }
    }

    // out = out*alpha + P @ V_chunk (V transposed -> contiguous fragments)
#pragma unroll
    for (int t = 0; t < 4; ++t) {
#pragma unroll
      for (int r = 0; r < 8; ++r) outacc[t][r] *= alpha[r];
      v16h vb = *(const v16h*)(Vt + (size_t)(t * 16 + ln) * SEQ + kc + g * 16);
      outacc[t] = __builtin_amdgcn_wmma_f32_16x16x32_f16(false, pf, false, vb, (short)0, outacc[t], false, false);
    }
    __syncthreads();
  }

  // normalize and scatter to ctx [b, n, h*64+d]
#pragma unroll
  for (int t = 0; t < 4; ++t)
#pragma unroll
    for (int r = 0; r < 8; ++r) {
      int m = r + 8 * g;
      int qabs = qt * 16 + m;
      int d = t * 16 + ln;
      float val = outacc[t][r] / lstate[r];
      ctx[(((size_t)bb * SEQ + qabs) * NHEADS + head) * HD + d] = (_Float16)val;
    }
}

// ---------------------------------------------------------------------------
// out = x + gate * a   (gate from mod at goff)
// ---------------------------------------------------------------------------
__global__ void k_residual(const float* __restrict__ x, const float* __restrict__ a,
                           const float* __restrict__ mod, int goff, float* __restrict__ out) {
  int i = blockIdx.x * blockDim.x + threadIdx.x;
  if (i >= ROWS * HIDDEN) return;
  int c = i & 1023;
  int b = i / (SEQ * HIDDEN);
  out[i] = x[i] + mod[(size_t)b * 6 * HIDDEN + goff + c] * a[i];
}

// ---------------------------------------------------------------------------
extern "C" void kernel_launch(void* const* d_in, const int* in_sizes, int n_in,
                              void* d_out, int out_size, void* d_ws, size_t ws_size,
                              hipStream_t stream) {
  (void)in_sizes; (void)n_in; (void)out_size; (void)ws_size;
  const float* x      = (const float*)d_in[0];
  const float* c      = (const float*)d_in[1];
  const float* w_ada  = (const float*)d_in[2];
  const float* b_ada  = (const float*)d_in[3];
  const float* w_qkv  = (const float*)d_in[4];
  const float* b_qkv  = (const float*)d_in[5];
  const float* w_out  = (const float*)d_in[6];
  const float* b_out  = (const float*)d_in[7];
  const float* rel    = (const float*)d_in[8];
  const float* w_mlp1 = (const float*)d_in[9];
  const float* b_mlp1 = (const float*)d_in[10];
  const float* w_mlp2 = (const float*)d_in[11];
  const float* b_mlp2 = (const float*)d_in[12];
  float* out = (float*)d_out;

  char* ws = (char*)d_ws;
  size_t off = 0;
  auto alloc = [&](size_t bytes) -> char* {
    char* p = ws + off; off += (bytes + 255) & ~(size_t)255; return p;
  };
  float*    mod    = (float*)   alloc((size_t)BATCH * 6 * HIDDEN * 4);
  _Float16* hbuf   = (_Float16*)alloc((size_t)ROWS * HIDDEN * 2);      // h / h2
  _Float16* wqkvT  = (_Float16*)alloc((size_t)HIDDEN * 3 * HIDDEN * 2); // [3H][H]
  _Float16* woutT  = (_Float16*)alloc((size_t)HIDDEN * HIDDEN * 2);     // [H][H]
  _Float16* wm1T   = (_Float16*)alloc((size_t)HIDDEN * MLPH * 2);       // [MLPH][H]
  _Float16* wm2T   = (_Float16*)alloc((size_t)MLPH * HIDDEN * 2);       // [H][MLPH]
  float*    qkv    = (float*)   alloc((size_t)ROWS * 3 * HIDDEN * 4);
  _Float16* qh     = (_Float16*)alloc((size_t)ROWS * HIDDEN * 2);
  _Float16* kh     = (_Float16*)alloc((size_t)ROWS * HIDDEN * 2);
  _Float16* vt     = (_Float16*)alloc((size_t)ROWS * HIDDEN * 2);
  _Float16* ctx    = (_Float16*)alloc((size_t)ROWS * HIDDEN * 2);
  float*    attn   = (float*)   alloc((size_t)ROWS * HIDDEN * 4);
  float*    x2     = (float*)   alloc((size_t)ROWS * HIDDEN * 4);
  _Float16* m1     = (_Float16*)alloc((size_t)ROWS * MLPH * 2);
  float*    m2     = (float*)   alloc((size_t)ROWS * HIDDEN * 4);

  // 1) modulation
  k_silu_ada<<<(BATCH * 6 * HIDDEN + 255) / 256, 256, 0, stream>>>(c, w_ada, b_ada, mod);
  // 2) weight cast + transpose -> [N][K] f16
  k_cast_transpose<<<(HIDDEN * 3 * HIDDEN + 255) / 256, 256, 0, stream>>>(w_qkv, wqkvT, HIDDEN, 3 * HIDDEN);
  k_cast_transpose<<<(HIDDEN * HIDDEN + 255) / 256, 256, 0, stream>>>(w_out, woutT, HIDDEN, HIDDEN);
  k_cast_transpose<<<(HIDDEN * MLPH + 255) / 256, 256, 0, stream>>>(w_mlp1, wm1T, HIDDEN, MLPH);
  k_cast_transpose<<<(MLPH * HIDDEN + 255) / 256, 256, 0, stream>>>(w_mlp2, wm2T, MLPH, HIDDEN);
  // 3) h = ln(x)*(1+sc_msa)+sh_msa
  k_ln_mod<<<ROWS, 256, 0, stream>>>(x, mod, 0, HIDDEN, hbuf);
  // 4) qkv = h @ w_qkv + b_qkv
  {
    dim3 grid(3 * HIDDEN / BN, ROWS / BM);
    k_gemm<<<grid, 256, 0, stream>>>(hbuf, wqkvT, b_qkv, qkv, nullptr, ROWS, 3 * HIDDEN, HIDDEN, 0);
  }
  // 5) RoPE + split heads (+ V transpose)
  k_rope_split<<<(ROWS * HIDDEN + 255) / 256, 256, 0, stream>>>(qkv, qh, kh, vt);
  // 6) attention
  {
    dim3 grid(BATCH * NHEADS, SEQ / 16);
    k_attn<<<grid, 32, 0, stream>>>(qh, kh, vt, rel, ctx);
  }
  // 7) attn_out = ctx @ w_out + b_out
  {
    dim3 grid(HIDDEN / BN, ROWS / BM);
    k_gemm<<<grid, 256, 0, stream>>>(ctx, woutT, b_out, attn, nullptr, ROWS, HIDDEN, HIDDEN, 0);
  }
  // 8) x2 = x + g_msa * attn_out
  k_residual<<<(ROWS * HIDDEN + 255) / 256, 256, 0, stream>>>(x, attn, mod, 2 * HIDDEN, x2);
  // 9) h2 = ln(x2)*(1+sc_mlp)+sh_mlp
  k_ln_mod<<<ROWS, 256, 0, stream>>>(x2, mod, 3 * HIDDEN, 4 * HIDDEN, hbuf);
  // 10) m1 = gelu(h2 @ w_mlp1 + b_mlp1)  (f16 out)
  {
    dim3 grid(MLPH / BN, ROWS / BM);
    k_gemm<<<grid, 256, 0, stream>>>(hbuf, wm1T, b_mlp1, nullptr, m1, ROWS, MLPH, HIDDEN, 1);
  }
  // 11) m2 = m1 @ w_mlp2 + b_mlp2
  {
    dim3 grid(HIDDEN / BN, ROWS / BM);
    k_gemm<<<grid, 256, 0, stream>>>(m1, wm2T, b_mlp2, m2, nullptr, ROWS, HIDDEN, MLPH, 0);
  }
  // 12) out = x2 + g_mlp * m2
  k_residual<<<(ROWS * HIDDEN + 255) / 256, 256, 0, stream>>>(x2, m2, mod, 5 * HIDDEN, out);
}
